// Mamba3D_81415400063085
// MI455X (gfx1250) — matrix operand order repaired and checked by reading, not compiled, not run on hardware
//
#include <hip/hip_runtime.h>
#include <hip/hip_bf16.h>
#include <math.h>

// ---------------------------------------------------------------------------
// Mamba3D forward for gfx1250 (MI455X).
// GEMMs: v_wmma_f32_16x16x32_f16 with packed-f16 operands (2x b128 loads per
// fragment). Selective scan: fp32 recurrence, LDS-staged tiles, coalesced
// reads, global_prefetch of the next chunk.
// ---------------------------------------------------------------------------

typedef __attribute__((ext_vector_type(16))) _Float16 v16h;
typedef __attribute__((ext_vector_type(8)))  float    v8f;

__device__ __forceinline__ v8f wmma_f32_f16(v16h a, v16h b, v8f c) {
  return __builtin_amdgcn_wmma_f32_16x16x32_f16(false, a, false, b,
                                                (short)0, c, false, false);
}

// Fragment loaders: packed-f16 rows, 16B-aligned, ISA 7.12.2 lane layouts.
union FragH { v16h v; float4 q[2]; };
__device__ __forceinline__ v16h ldA(const _Float16* __restrict__ row, int kk, int half) {
  FragH f;
  f.q[0] = *(const float4*)(row + kk + half * 8);        // e=0..7 : k=kk+half*8+e
  f.q[1] = *(const float4*)(row + kk + 16 + half * 8);   // e=8..15: k=kk+16+half*8+(e-8)
  return f.v;
}
__device__ __forceinline__ v16h ldB(const _Float16* __restrict__ row, int kk, int half) {
  FragH f;
  f.q[0] = *(const float4*)(row + kk + half * 8);        // e=0..7 : k=kk+half*8+e
  f.q[1] = *(const float4*)(row + kk + 16 + half * 8);   // e=8..15: k=kk+16+half*8+(e-8)
  return f.v;
}

// Problem constants
#define NTOK   4096
#define DM     128
#define DI     192
#define DH     96
#define DTR    8
#define NST    16
#define LSEQ   2048
#define NGRP   16

// ---- f32 workspace offsets (floats) ----
#define OFF_XZ    0u           // 4096*384
#define OFF_XC    1572864u     // 2*2048*192   layout [b][sp][c]
#define OFF_XDBL  2359296u     // 16*40*2048
#define OFF_Y4    3670016u     // 2*4*4096*96
#define OFF_T1    6815744u     // 2*4096*192
#define OFF_WTS   8388608u     // 2*4096*4
#define NF32      8421376u     // total floats
// ---- f16 workspace offsets (halves), region starts at NF32 floats ----
#define HOFF_XH    0u          // 4096*128
#define HOFF_WIH   524288u     // 384*128
#define HOFF_XPWH  573440u     // 2*4*48*96 (rows zero-padded 40->48)
#define HOFF_FW1H  610304u     // 2*192*384
#define HOFF_OPWH  757760u     // 128*192
#define HOFF_XSH   782336u     // 16*2048*96  [g][l][c]
#define HOFF_XCATH 3928064u    // 2*4096*384  [s][tok][j*96+c]
#define HOFF_GH    7073792u    // 4096*192

__device__ __forceinline__ int dir_of(int s, int k) {
  const int d1[4] = {2, 3, 6, 7};   // core0: DIRS1
  const int d2[4] = {0, 1, 4, 5};   // core1: DIRS2
  return s ? d2[k] : d1[k];
}

// sequence position l -> spatial index sp = f*256 + h*16 + w
__device__ __forceinline__ int seq_sp(int p, int fl, int l) {
  int lq = fl ? (2047 - l) : l;
  switch (p) {
    case 0:  return lq;
    case 1:  return (lq & 0x700) | ((lq & 15) << 4) | ((lq >> 4) & 15);
    case 2:  return ((lq & 7) << 8) | (((lq >> 7) & 15) << 4) | ((lq >> 3) & 15);
    default: return ((lq & 7) << 8) | (((lq >> 3) & 15) << 4) | ((lq >> 7) & 15);
  }
}

// ---------------------------------------------------------------------------
// Pack helpers: f32 -> f16
// ---------------------------------------------------------------------------
__global__ void k_pack(const float* __restrict__ src, _Float16* __restrict__ dst, int n) {
  int i = blockIdx.x * blockDim.x + threadIdx.x;
  if (i < n) dst[i] = (_Float16)src[i];
}

__global__ void k_pack_xpw(const float* __restrict__ A, const float* __restrict__ B,
                           _Float16* __restrict__ dst) {
  int i = blockIdx.x * blockDim.x + threadIdx.x;     // over 2*4*48*96
  if (i >= 2 * 4 * 48 * 96) return;
  int c = i % 96, r = (i / 96) % 48, k = (i / (96 * 48)) % 4, s = i / (96 * 48 * 4);
  const float* src = s ? B : A;
  dst[i] = (r < 40) ? (_Float16)src[(k * 40 + r) * 96 + c] : (_Float16)0.f;
}

// ---------------------------------------------------------------------------
// K1: xz = x @ in_proj_w^T  (WMMA, packed f16 operands)
// ---------------------------------------------------------------------------
__global__ void k_gemm_inproj(const _Float16* __restrict__ XH,
                              const _Float16* __restrict__ WIH,
                              float* __restrict__ XZ) {
  int mt = blockIdx.x, nt = blockIdx.y;
  int lane = threadIdx.x & 31, half = lane >> 4, lm = lane & 15;
  const _Float16* arow = XH + (size_t)(mt * 16 + lm) * DM;
  const _Float16* brow = WIH + (size_t)(nt * 16 + lm) * DM;
  v8f acc = {};
#pragma unroll
  for (int kk = 0; kk < 128; kk += 32)
    acc = wmma_f32_f16(ldA(arow, kk, half), ldB(brow, kk, half), acc);
#pragma unroll
  for (int v = 0; v < 8; ++v)
    XZ[(size_t)(mt * 16 + v + 8 * half) * 384 + nt * 16 + lm] = acc[v];
}

// ---------------------------------------------------------------------------
// K2: depthwise 3x3x3 conv + bias + SiLU -> XC[(b*2048+sp)*192 + c]
// ---------------------------------------------------------------------------
__global__ void k_conv_silu(const float* __restrict__ XZ,
                            const float* __restrict__ CW,
                            const float* __restrict__ CB,
                            float* __restrict__ XC) {
  int idx = blockIdx.x * blockDim.x + threadIdx.x;
  if (idx >= 2 * LSEQ * DI) return;
  int c  = idx % DI;
  int sp = (idx / DI) & 2047;
  int b  = idx / (DI * LSEQ);
  int f = sp >> 8, h = (sp >> 4) & 15, w = sp & 15;
  float acc = CB[c];
#pragma unroll
  for (int df = -1; df <= 1; ++df) {
    int ff = f + df; if ((unsigned)ff >= 8u) continue;
#pragma unroll
    for (int dh = -1; dh <= 1; ++dh) {
      int hh = h + dh; if ((unsigned)hh >= 16u) continue;
#pragma unroll
      for (int dw = -1; dw <= 1; ++dw) {
        int ww = w + dw; if ((unsigned)ww >= 16u) continue;
        float xin = XZ[(size_t)((b * 8 + ff) * 256 + hh * 16 + ww) * 384 + c];
        acc += xin * CW[c * 27 + (df + 1) * 9 + (dh + 1) * 3 + (dw + 1)];
      }
    }
  }
  XC[idx] = acc / (1.f + __expf(-acc));
}

// ---------------------------------------------------------------------------
// K2b: build xs (f16) in sequence order: XSH[(g*2048+l)*96 + c]
// ---------------------------------------------------------------------------
__global__ void k_build_xs(const float* __restrict__ XC, _Float16* __restrict__ XSH) {
  int idx = blockIdx.x * blockDim.x + threadIdx.x;    // 16*2048*96
  if (idx >= NGRP * LSEQ * DH) return;
  int c = idx % DH;
  int l = (idx / DH) & 2047;
  int g = idx / (DH * LSEQ);
  int s = g >> 3, b = (g >> 2) & 1, k = g & 3;
  int dir = dir_of(s, k), p = dir & 3, fl = dir >> 2;
  int sp = seq_sp(p, fl, l);
  XSH[idx] = (_Float16)XC[(size_t)(b * LSEQ + sp) * DI + s * DH + c];
}

// ---------------------------------------------------------------------------
// K3: xdbl[g] = x_proj_w[k] (48-pad x 96) @ xs (96 x 2048)   (WMMA)
// ---------------------------------------------------------------------------
__global__ void k_gemm_xdbl(const _Float16* __restrict__ XPWH,
                            const _Float16* __restrict__ XSH,
                            float* __restrict__ XDBL) {
  int mt = blockIdx.x;            // 0..2
  int l0 = blockIdx.y * 16;
  int g  = blockIdx.z;
  int s = g >> 3, k = g & 3;
  int lane = threadIdx.x & 31, half = lane >> 4, lm = lane & 15;
  const _Float16* arow = XPWH + (size_t)((s * 4 + k) * 48 + mt * 16 + lm) * DH;
  const _Float16* brow = XSH + (size_t)(g * LSEQ + l0 + lm) * DH;
  v8f acc = {};
#pragma unroll
  for (int kk = 0; kk < 96; kk += 32)
    acc = wmma_f32_f16(ldA(arow, kk, half), ldB(brow, kk, half), acc);
#pragma unroll
  for (int v = 0; v < 8; ++v) {
    int r = mt * 16 + v + 8 * half;
    if (r < 40) XDBL[(size_t)g * 40 * LSEQ + (size_t)r * LSEQ + l0 + lm] = acc[v];
  }
}

// ---------------------------------------------------------------------------
// K4: selective scan. One block per group, lane = channel c (0..95).
// Writes fp32 Y4 (restored layout) AND f16 xcat for the fusion GEMM.
// ---------------------------------------------------------------------------
__global__ void k_scan(const float* __restrict__ XC,
                       const float* __restrict__ XDBL,
                       const float* __restrict__ DTW1, const float* __restrict__ DTB1,
                       const float* __restrict__ AL1,  const float* __restrict__ DS1,
                       const float* __restrict__ DTW2, const float* __restrict__ DTB2,
                       const float* __restrict__ AL2,  const float* __restrict__ DS2,
                       float* __restrict__ Y4, _Float16* __restrict__ XCATH) {
  __shared__ float ld[40][65];
  int g = blockIdx.x;
  int s = g >> 3, b = (g >> 2) & 1, k = g & 3;
  int dir = dir_of(s, k), p = dir & 3, fl = dir >> 2;
  int c = threadIdx.x;                         // 0..95
  const float* DTW = s ? DTW2 : DTW1;
  const float* DTB = s ? DTB2 : DTB1;
  const float* AL  = s ? AL2  : AL1;
  const float* DS  = s ? DS2  : DS1;

  float dtw[DTR];
#pragma unroll
  for (int r = 0; r < DTR; ++r) dtw[r] = DTW[(k * DH + c) * DTR + r];
  float dtb = DTB[k * DH + c];
  float Ac[NST];
#pragma unroll
  for (int n = 0; n < NST; ++n) Ac[n] = -__expf(AL[(k * DH + c) * NST + n]);
  float dsc = DS[k * DH + c];

  const float* xcb = XC + (size_t)b * LSEQ * DI + s * DH + c;   // + sp*DI
  const float* xd  = XDBL + (size_t)g * 40 * LSEQ;
  float* y4o = Y4 + (size_t)s * 4 * NTOK * DH + (size_t)k * NTOK * DH
                  + (size_t)b * LSEQ * DH;
  _Float16* xco = XCATH + (size_t)s * NTOK * 384 + (size_t)b * LSEQ * 384 + k * DH + c;

  float h[NST];
#pragma unroll
  for (int n = 0; n < NST; ++n) h[n] = 0.f;

  for (int l0 = 0; l0 < LSEQ; l0 += 64) {
    __syncthreads();
    for (int idx = threadIdx.x; idx < 40 * 64; idx += 96) {
      int r = idx >> 6, t = idx & 63;
      ld[r][t] = xd[(size_t)r * LSEQ + l0 + t];
      // prefetch next chunk (global_prefetch_b8) while recurrence runs
      if (((idx & 31) == 0) && (l0 + 64 < LSEQ))
        __builtin_prefetch(&xd[(size_t)r * LSEQ + l0 + 64 + t], 0, 1);
    }
    __syncthreads();
    for (int t = 0; t < 64; ++t) {
      int l = l0 + t;
      float dts = dtb;
#pragma unroll
      for (int r = 0; r < DTR; ++r) dts += dtw[r] * ld[r][t];
      float delta = (dts > 20.f) ? dts : log1pf(__expf(dts));   // softplus
      int sp = seq_sp(p, fl, l);
      float u  = xcb[(size_t)sp * DI];
      float du = delta * u;
      float y  = 0.f;
#pragma unroll
      for (int n = 0; n < NST; ++n) {
        float a = __expf(delta * Ac[n]);
        h[n] = a * h[n] + du * ld[8 + n][t];
        y   += h[n] * ld[24 + n][t];
      }
      float out = y + dsc * u;
      y4o[(size_t)sp * DH + c] = out;
      xco[(size_t)sp * 384]    = (_Float16)out;
    }
  }
}

// ---------------------------------------------------------------------------
// K5a: t1 = relu(xcat @ fw1^T + fb1)   (WMMA, table-order f16 fragments)
// ---------------------------------------------------------------------------
__global__ void k_gemm_fus1(const _Float16* __restrict__ XCATH,
                            const _Float16* __restrict__ FW1H,
                            const float* __restrict__ FB1a, const float* __restrict__ FB1b,
                            float* __restrict__ T1) {
  int mt = blockIdx.x, nt = blockIdx.y, s = blockIdx.z;
  const float* FB1 = s ? FB1b : FB1a;
  int lane = threadIdx.x & 31, half = lane >> 4, lm = lane & 15;
  const _Float16* arow = XCATH + (size_t)s * NTOK * 384 + (size_t)(mt * 16 + lm) * 384;
  const _Float16* brow = FW1H + (size_t)s * DI * 384 + (size_t)(nt * 16 + lm) * 384;
  v8f acc = {};
#pragma unroll
  for (int kk = 0; kk < 384; kk += 32)
    acc = wmma_f32_f16(ldA(arow, kk, half), ldB(brow, kk, half), acc);
#pragma unroll
  for (int v = 0; v < 8; ++v) {
    int m = mt * 16 + v + 8 * half, o = nt * 16 + lm;
    float val = acc[v] + FB1[o];
    T1[(size_t)s * NTOK * DI + (size_t)m * DI + o] = val > 0.f ? val : 0.f;
  }
}

// ---------------------------------------------------------------------------
// K5b: per-token direction weights = softmax(t1 @ fw2^T + fb2) over 4 dirs
// ---------------------------------------------------------------------------
__global__ void k_fus_w(const float* __restrict__ T1,
                        const float* __restrict__ FW2a, const float* __restrict__ FB2a,
                        const float* __restrict__ FW2b, const float* __restrict__ FB2b,
                        float* __restrict__ WTS) {
  int idx = blockIdx.x * blockDim.x + threadIdx.x;
  if (idx >= 2 * NTOK) return;
  int s = idx >> 12, tok = idx & (NTOK - 1);
  const float* FW2 = s ? FW2b : FW2a;
  const float* FB2 = s ? FB2b : FB2a;
  const float* t1 = T1 + (size_t)s * NTOK * DI + (size_t)tok * DI;
  float lg[4];
#pragma unroll
  for (int j = 0; j < 4; ++j) {
    float acc = FB2[j];
    for (int o = 0; o < DI; ++o) acc += t1[o] * FW2[j * DI + o];
    lg[j] = acc;
  }
  float mx = fmaxf(fmaxf(lg[0], lg[1]), fmaxf(lg[2], lg[3]));
  float ex[4], sum = 0.f;
#pragma unroll
  for (int j = 0; j < 4; ++j) { ex[j] = __expf(lg[j] - mx); sum += ex[j]; }
  float inv = 1.f / sum;
#pragma unroll
  for (int j = 0; j < 4; ++j) WTS[(size_t)idx * 4 + j] = ex[j] * inv;
}

// ---------------------------------------------------------------------------
// K6a: weighted combine + LayerNorm(192) + SiLU gate -> GH (f16)
// ---------------------------------------------------------------------------
__global__ void k_ln_gate(const float* __restrict__ Y4,
                          const float* __restrict__ WTS,
                          const float* __restrict__ XZ,
                          const float* __restrict__ LNG,
                          const float* __restrict__ LNB,
                          _Float16* __restrict__ GH) {
  __shared__ float sv[DI];
  int tok = blockIdx.x, c = threadIdx.x;
  int s = (c >= DH) ? 1 : 0, cc = c - s * DH;
  const float* y4 = Y4 + (size_t)s * 4 * NTOK * DH;
  const float* wt = WTS + (size_t)s * NTOK * 4 + (size_t)tok * 4;
  float val = 0.f;
#pragma unroll
  for (int j = 0; j < 4; ++j)
    val += y4[((size_t)j * NTOK + tok) * DH + cc] * wt[j];
  sv[c] = val;
  __syncthreads();
  float sum = 0.f, sq = 0.f;
  for (int i = 0; i < DI; ++i) { float t = sv[i]; sum += t; sq += t * t; }
  float mu  = sum * (1.f / DI);
  float var = sq * (1.f / DI) - mu * mu;
  float rstd = rsqrtf(var + 1e-5f);
  float yn = (val - mu) * rstd * LNG[c] + LNB[c];
  float z  = XZ[(size_t)tok * 384 + DI + c];
  float sz = z / (1.f + __expf(-z));
  GH[(size_t)tok * DI + c] = (_Float16)(yn * sz);
}

// ---------------------------------------------------------------------------
// K6b: out = G @ out_proj_w^T   (WMMA) -> d_out (f32)
// ---------------------------------------------------------------------------
__global__ void k_gemm_out(const _Float16* __restrict__ GH,
                           const _Float16* __restrict__ OPWH,
                           float* __restrict__ OUT) {
  int mt = blockIdx.x, nt = blockIdx.y;
  int lane = threadIdx.x & 31, half = lane >> 4, lm = lane & 15;
  const _Float16* arow = GH + (size_t)(mt * 16 + lm) * DI;
  const _Float16* brow = OPWH + (size_t)(nt * 16 + lm) * DI;
  v8f acc = {};
#pragma unroll
  for (int kk = 0; kk < 192; kk += 32)
    acc = wmma_f32_f16(ldA(arow, kk, half), ldB(brow, kk, half), acc);
#pragma unroll
  for (int v = 0; v < 8; ++v)
    OUT[(size_t)(mt * 16 + v + 8 * half) * DM + nt * 16 + lm] = acc[v];
}

// ---------------------------------------------------------------------------
extern "C" void kernel_launch(void* const* d_in, const int* in_sizes, int n_in,
                              void* d_out, int out_size, void* d_ws, size_t ws_size,
                              hipStream_t stream) {
  const float* x        = (const float*)d_in[0];
  const float* in_proj  = (const float*)d_in[1];
  const float* conv_w   = (const float*)d_in[2];
  const float* conv_b   = (const float*)d_in[3];
  const float* xpw1     = (const float*)d_in[4];
  const float* dtw1     = (const float*)d_in[5];
  const float* dtb1     = (const float*)d_in[6];
  const float* al1      = (const float*)d_in[7];
  const float* ds1      = (const float*)d_in[8];
  const float* fw1_1    = (const float*)d_in[9];
  const float* fb1_1    = (const float*)d_in[10];
  const float* fw2_1    = (const float*)d_in[11];
  const float* fb2_1    = (const float*)d_in[12];
  const float* xpw2     = (const float*)d_in[13];
  const float* dtw2     = (const float*)d_in[14];
  const float* dtb2     = (const float*)d_in[15];
  const float* al2      = (const float*)d_in[16];
  const float* ds2      = (const float*)d_in[17];
  const float* fw1_2    = (const float*)d_in[18];
  const float* fb1_2    = (const float*)d_in[19];
  const float* fw2_2    = (const float*)d_in[20];
  const float* fb2_2    = (const float*)d_in[21];
  const float* ln_g     = (const float*)d_in[22];
  const float* ln_b     = (const float*)d_in[23];
  const float* out_proj = (const float*)d_in[24];

  float* ws   = (float*)d_ws;
  float* XZ   = ws + OFF_XZ;
  float* XC   = ws + OFF_XC;
  float* XDBL = ws + OFF_XDBL;
  float* Y4   = ws + OFF_Y4;
  float* T1   = ws + OFF_T1;
  float* WTS  = ws + OFF_WTS;
  _Float16* hws   = (_Float16*)(ws + NF32);
  _Float16* XH    = hws + HOFF_XH;
  _Float16* WIH   = hws + HOFF_WIH;
  _Float16* XPWH  = hws + HOFF_XPWH;
  _Float16* FW1H  = hws + HOFF_FW1H;
  _Float16* OPWH  = hws + HOFF_OPWH;
  _Float16* XSH   = hws + HOFF_XSH;
  _Float16* XCATH = hws + HOFF_XCATH;
  _Float16* GH    = hws + HOFF_GH;
  float* OUT = (float*)d_out;

  // 0) pack f16 operands (activations x + all GEMM weights)
  k_pack<<<(4096 * 128 + 255) / 256, 256, 0, stream>>>(x, XH, 4096 * 128);
  k_pack<<<(384 * 128 + 255) / 256, 256, 0, stream>>>(in_proj, WIH, 384 * 128);
  k_pack<<<(192 * 384 + 255) / 256, 256, 0, stream>>>(fw1_1, FW1H, 192 * 384);
  k_pack<<<(192 * 384 + 255) / 256, 256, 0, stream>>>(fw1_2, FW1H + 192 * 384, 192 * 384);
  k_pack<<<(128 * 192 + 255) / 256, 256, 0, stream>>>(out_proj, OPWH, 128 * 192);
  k_pack_xpw<<<(2 * 4 * 48 * 96 + 255) / 256, 256, 0, stream>>>(xpw1, xpw2, XPWH);

  // 1) in_proj GEMM (WMMA)
  k_gemm_inproj<<<dim3(NTOK / 16, 384 / 16), 32, 0, stream>>>(XH, WIH, XZ);
  // 2) depthwise conv + SiLU (spatial-major layout), then xs (f16) builder
  k_conv_silu<<<(2 * LSEQ * DI + 255) / 256, 256, 0, stream>>>(XZ, conv_w, conv_b, XC);
  k_build_xs<<<(NGRP * LSEQ * DH + 255) / 256, 256, 0, stream>>>(XC, XSH);
  // 3) per-direction x-projection GEMMs (WMMA)
  k_gemm_xdbl<<<dim3(3, LSEQ / 16, NGRP), 32, 0, stream>>>(XPWH, XSH, XDBL);
  // 4) selective scan
  k_scan<<<NGRP, 96, 0, stream>>>(XC, XDBL, dtw1, dtb1, al1, ds1,
                                  dtw2, dtb2, al2, ds2, Y4, XCATH);
  // 5) fusion MLP layer 1 (WMMA) + direction softmax weights
  k_gemm_fus1<<<dim3(NTOK / 16, DI / 16, 2), 32, 0, stream>>>(XCATH, FW1H,
                                                              fb1_1, fb1_2, T1);
  k_fus_w<<<(2 * NTOK + 127) / 128, 128, 0, stream>>>(T1, fw2_1, fb2_1,
                                                      fw2_2, fb2_2, WTS);
  // 6) combine + LayerNorm + SiLU gate (f16 out), then out_proj GEMM (WMMA)
  k_ln_gate<<<NTOK, DI, 0, stream>>>(Y4, WTS, XZ, ln_g, ln_b, GH);
  k_gemm_out<<<dim3(NTOK / 16, DM / 16), 32, 0, stream>>>(GH, OPWH, OUT);
}
